// MPNN_17162689315346
// MI455X (gfx1250) — matrix operand order repaired and checked by reading, compiled, not run on hardware
//
#include <hip/hip_runtime.h>
#include <hip/hip_bf16.h>
#include <math.h>

// ---------------- problem constants ----------------
#define NN    25000
#define EE    100000
#define FIN   16
#define HH1   32
#define HH2   64
#define NN1   12500
#define NN2   6250
#define BB    64
#define CC    10

#define K1PAD 416   // 25*16 edge-weight rows + 16 bias rows (exact)
#define K2PAD 832   // 25*32 edge-weight rows + 32 bias rows (exact)

typedef __attribute__((ext_vector_type(16))) _Float16 v16h;
typedef __attribute__((ext_vector_type(8)))  float    v8f;

// ---------------- small helpers ----------------
__device__ inline void atomicMaxF(float* addr, float v) {
  // ordered-int trick: valid for mixed-sign floats
  if (v >= 0.0f) atomicMax((int*)addr, __float_as_int(v));
  else           atomicMin((unsigned int*)addr, __float_as_uint(v));
}

__global__ void fill_f32(float* p, float v, int n) {
  int i = blockIdx.x * blockDim.x + threadIdx.x;
  if (i < n) p[i] = v;
}
__global__ void fill_i32(int* p, int v, int n) {
  int i = blockIdx.x * blockDim.x + threadIdx.x;
  if (i < n) p[i] = v;
}

// ---------------- B-matrix conversion: f32 -> f16 in WMMA *fragment order* --
// Fragment order: Bh[ ((kt*NT + t)*32 + lane)*16 + el ], el = 2v+j,
// logical element B[kk][o] with kk = kt*32 + 16*(lane>>4) + el,
// o = t*16 + (lane&15). Logical B: kk = k*F+i -> wb[k, i*H+o];
// kk >= 25*F -> folded edge-bias rows bb[i*H+o].
template<int F, int H, int KPAD>
__global__ void makeBfrag(const float* __restrict__ wb, const float* __restrict__ bb,
                          _Float16* __restrict__ Bh) {
  constexpr int NT = H / 16;
  constexpr int TOTAL = (KPAD / 32) * NT * 512;
  int idx = blockIdx.x * blockDim.x + threadIdx.x;
  if (idx >= TOTAL) return;
  int el   = idx & 15;
  int lane = (idx >> 4) & 31;
  int rem  = idx >> 9;
  int t    = rem % NT;
  int kt   = rem / NT;
  int kk = kt * 32 + 16 * (lane >> 4) + el;
  int o  = t * 16 + (lane & 15);
  float v;
  if (kk < 25 * F) { int k = kk / F, i = kk % F; v = wb[k * (F * H) + i * H + o]; }
  else             { int i = kk - 25 * F;        v = bb[i * H + o]; }
  Bh[idx] = (_Float16)v;
}

// fc1_w [64,128] f32 -> fragment-order f16 (K=64 -> 2 K-tiles, N=128 -> 8 N-tiles)
__global__ void makeBfrag_fc1(const float* __restrict__ w, _Float16* __restrict__ Bh) {
  int idx = blockIdx.x * blockDim.x + threadIdx.x;
  if (idx >= 2 * 8 * 512) return;
  int el   = idx & 15;
  int lane = (idx >> 4) & 31;
  int rem  = idx >> 9;
  int t    = rem & 7;
  int kt   = rem >> 3;
  int kk = kt * 32 + 16 * (lane >> 4) + el;
  int o  = t * 16 + (lane & 15);
  Bh[idx] = (_Float16)w[kk * 128 + o];
}

// ---------------- fused NNConv message + scatter (WMMA) ----------------
// One wave handles 16 edges. A-tile [16, KPAD] = outer(relu(ea@wa+ba), x_src)
// built in LDS (f16) using both half-waves; GEMM vs pre-swizzled B with
// v_wmma_f32_16x16x32_f16; f32 C tiles scattered via atomicAdd.
template<int F, int H, int KPAD, bool SECOND>
__global__ __launch_bounds__(32)
void nnconv_msg(const float* __restrict__ xfeat,       // [Nsrc, F]
                const float* __restrict__ ea_or_pos,   // edge_attr[E,2] or pos_p[N1,2]
                const float* __restrict__ wa,          // [2,25]
                const float* __restrict__ ba,          // [25]
                const int*   __restrict__ edge_index,  // [2,E]
                const int*   __restrict__ cluster1,    // [N] (SECOND only)
                const float* __restrict__ maxabs,      // [1]  (SECOND only)
                const _Float16* __restrict__ Bh,       // fragment-order
                float* __restrict__ sacc,              // [Ndst, H]
                float* __restrict__ cnt)               // [Ndst]
{
  __shared__ _Float16 A[16][KPAD];
  const int lane = threadIdx.x;
  const int e0 = blockIdx.x * 16;
  const int m  = lane & 15;     // edge within tile
  const int kh = lane >> 4;     // which half of the k-range this lane builds

  {
    const int e = e0 + m;
    int src;
    float ea0, ea1;
    if (!SECOND) {
      src = edge_index[e];
      ea0 = ea_or_pos[2 * e];
      ea1 = ea_or_pos[2 * e + 1];
    } else {
      src = cluster1[edge_index[e]];
      int dst = cluster1[edge_index[EE + e]];
      float inv = 1.0f / (2.0f * maxabs[0]);
      ea0 = (ea_or_pos[2 * dst]     - ea_or_pos[2 * src])     * inv + 0.5f;
      ea1 = (ea_or_pos[2 * dst + 1] - ea_or_pos[2 * src + 1]) * inv + 0.5f;
    }
    float u[F];
    #pragma unroll
    for (int i = 0; i < F; i++) u[i] = xfeat[src * F + i];
    const int kbeg = kh ? 13 : 0;
    const int kend = kh ? 25 : 13;
    #pragma unroll 1
    for (int k = kbeg; k < kend; k++) {
      float h = fmaf(ea0, wa[k], fmaf(ea1, wa[25 + k], ba[k]));
      h = h > 0.0f ? h : 0.0f;
      #pragma unroll
      for (int i = 0; i < F; i++) A[m][k * F + i] = (_Float16)(h * u[i]);
    }
    if (kh) {
      #pragma unroll
      for (int i = 0; i < F; i++) A[m][25 * F + i] = (_Float16)u[i];  // bias rows
    }
  }
  __syncthreads();

  constexpr int NT = H / 16;
  v8f zero = {};
  v8f acc[NT];
  #pragma unroll
  for (int t = 0; t < NT; t++) acc[t] = zero;

  #pragma unroll 1
  for (int kt = 0; kt < KPAD / 32; kt++) {
    // A fragment: lane L holds M = L&15; K-pairs contiguous -> ds_load_b128 x2
    v16h a;
    #pragma unroll
    for (int v = 0; v < 8; v++) {
      const int kb = 2 * v + 16 * kh;
      a[2 * v]     = A[m][kt * 32 + kb];
      a[2 * v + 1] = A[m][kt * 32 + kb + 1];
    }
    #pragma unroll
    for (int t = 0; t < NT; t++) {
      // pre-swizzled: one aligned 32-byte load per fragment per lane
      const _Float16* bp = Bh + (size_t)((kt * NT + t) * 32 + lane) * 16;
      v16h b = *(const v16h*)bp;
      acc[t] = __builtin_amdgcn_wmma_f32_16x16x32_f16(
          false, a, false, b, (short)0, acc[t], false, false);
    }
  }

  // scatter: C lane L holds N = t*16 + (L&15); VGPR r -> edge m = r + 8*(L>>4)
  #pragma unroll 1
  for (int r = 0; r < 8; r++) {
    const int me = r + 8 * kh;
    int dst = edge_index[EE + e0 + me];
    if (SECOND) dst = cluster1[dst];
    #pragma unroll
    for (int t = 0; t < NT; t++)
      atomicAdd(&sacc[dst * H + t * 16 + m], acc[t][r]);
  }
  if (lane < 16) {
    int dst = edge_index[EE + e0 + lane];
    if (SECOND) dst = cluster1[dst];
    atomicAdd(&cnt[dst], 1.0f);
  }
}

// ---------------- node update: mean + root term + bias + ELU -------------
template<int F, int H>
__global__ void node_update(const float* __restrict__ s, const float* __restrict__ cnt,
                            const float* __restrict__ xin, const float* __restrict__ root,
                            const float* __restrict__ bias, float* __restrict__ xout, int n) {
  int idx = blockIdx.x * blockDim.x + threadIdx.x;
  if (idx >= n * H) return;
  int v = idx / H, o = idx % H;
  float r = 0.0f;
  #pragma unroll
  for (int i = 0; i < F; i++) r = fmaf(xin[v * F + i], root[i * H + o], r);
  float val = s[idx] / fmaxf(cnt[v], 1.0f) + r + bias[o];
  xout[idx] = val > 0.0f ? val : expm1f(val);
}

// ---------------- pooling kernels ----------------
__global__ void pool1(const float* __restrict__ x1, const float* __restrict__ pos,
                      const int* __restrict__ cluster1, const int* __restrict__ batch,
                      float* __restrict__ x_p, float* __restrict__ pos_sum,
                      float* __restrict__ cntp, int* __restrict__ batch2) {
  int idx = blockIdx.x * blockDim.x + threadIdx.x;
  if (idx >= NN * HH1) return;
  int v = idx / HH1, o = idx % HH1;
  int c = cluster1[v];
  atomicMaxF(&x_p[c * HH1 + o], x1[idx]);
  if (o == 0) {
    atomicAdd(&pos_sum[c * 2],     pos[v * 2]);
    atomicAdd(&pos_sum[c * 2 + 1], pos[v * 2 + 1]);
    atomicAdd(&cntp[c], 1.0f);
    atomicMax(&batch2[c], batch[v]);
  }
}

__global__ void posdiv(const float* __restrict__ pos_sum, const float* __restrict__ cntp,
                       float* __restrict__ pos_p) {
  int i = blockIdx.x * blockDim.x + threadIdx.x;
  if (i >= NN1) return;
  float c = fmaxf(cntp[i], 1.0f);
  pos_p[i * 2]     = pos_sum[i * 2] / c;
  pos_p[i * 2 + 1] = pos_sum[i * 2 + 1] / c;
}

__global__ void cart_maxabs(const float* __restrict__ pos_p, const int* __restrict__ edge_index,
                            const int* __restrict__ cluster1, unsigned int* __restrict__ mx) {
  __shared__ float red[256];
  int e = blockIdx.x * 256 + threadIdx.x;
  float m = 0.0f;
  if (e < EE) {
    int r = cluster1[edge_index[e]], c = cluster1[edge_index[EE + e]];
    float c0 = pos_p[c * 2]     - pos_p[r * 2];
    float c1 = pos_p[c * 2 + 1] - pos_p[r * 2 + 1];
    m = fmaxf(fabsf(c0), fabsf(c1));
  }
  red[threadIdx.x] = m;
  __syncthreads();
  for (int s = 128; s > 0; s >>= 1) {
    if (threadIdx.x < s) red[threadIdx.x] = fmaxf(red[threadIdx.x], red[threadIdx.x + s]);
    __syncthreads();
  }
  if (threadIdx.x == 0) atomicMax(mx, __float_as_uint(red[0]));  // nonneg: uint order ok
}

__global__ void pool2(const float* __restrict__ x2, const int* __restrict__ cluster2,
                      const int* __restrict__ batch2, float* __restrict__ x_p2,
                      int* __restrict__ batch3) {
  int idx = blockIdx.x * blockDim.x + threadIdx.x;
  if (idx >= NN1 * HH2) return;
  int v = idx / HH2, o = idx % HH2;
  int c = cluster2[v];
  atomicMaxF(&x_p2[c * HH2 + o], x2[idx]);
  if (o == 0) atomicMax(&batch3[c], batch2[v]);
}

__global__ void gpool(const float* __restrict__ x_p2, const int* __restrict__ batch3,
                      float* __restrict__ g, float* __restrict__ gc) {
  int idx = blockIdx.x * blockDim.x + threadIdx.x;
  if (idx >= NN2 * HH2) return;
  int v = idx / HH2, o = idx % HH2;
  int b = batch3[v];
  atomicAdd(&g[b * HH2 + o], x_p2[idx]);
  if (o == 0) atomicAdd(&gc[b], 1.0f);
}

__global__ void gdiv(float* __restrict__ g, const float* __restrict__ gc) {
  int idx = blockIdx.x * blockDim.x + threadIdx.x;
  if (idx >= BB * HH2) return;
  g[idx] /= fmaxf(gc[idx / HH2], 1.0f);
}

// ---------------- MLP head ----------------
// fc1 as WMMA: [64,64] @ [64,128], 4 waves = 4 M-tiles, 8 N-tiles, 2 K-steps.
__global__ __launch_bounds__(128)
void fc1_wmma(const float* __restrict__ g, const _Float16* __restrict__ Bh,
              const float* __restrict__ fcb, float* __restrict__ hfc) {
  __shared__ _Float16 G[64][64];
  const int tid = threadIdx.x;
  for (int i = tid; i < 64 * 64; i += 128) G[i >> 6][i & 63] = (_Float16)g[i];
  __syncthreads();

  const int wave = tid >> 5, lane = tid & 31;
  const int M = lane & 15, hh = lane >> 4;
  v8f zero = {};
  v8f acc[8];
  #pragma unroll
  for (int t = 0; t < 8; t++) acc[t] = zero;

  #pragma unroll
  for (int kt = 0; kt < 2; kt++) {
    v16h a;
    #pragma unroll
    for (int v = 0; v < 8; v++) {
      const int kb = 2 * v + 16 * hh;
      a[2 * v]     = G[wave * 16 + M][kt * 32 + kb];
      a[2 * v + 1] = G[wave * 16 + M][kt * 32 + kb + 1];
    }
    #pragma unroll
    for (int t = 0; t < 8; t++) {
      const _Float16* bp = Bh + (size_t)((kt * 8 + t) * 32 + lane) * 16;
      v16h b = *(const v16h*)bp;
      acc[t] = __builtin_amdgcn_wmma_f32_16x16x32_f16(
          false, a, false, b, (short)0, acc[t], false, false);
    }
  }

  #pragma unroll
  for (int r = 0; r < 8; r++) {
    const int row = wave * 16 + r + 8 * hh;
    #pragma unroll
    for (int t = 0; t < 8; t++) {
      const int col = t * 16 + M;
      float val = acc[t][r] + fcb[col];
      hfc[row * 128 + col] = val > 0.0f ? val : expm1f(val);
    }
  }
}

__global__ void fc2k(const float* __restrict__ h, const float* __restrict__ w,
                     const float* __restrict__ b, float* __restrict__ out) {
  int bb = threadIdx.x;
  if (bb >= BB) return;
  float l[CC];
  for (int c = 0; c < CC; c++) {
    float a = b[c];
    for (int k = 0; k < 128; k++) a = fmaf(h[bb * 128 + k], w[k * CC + c], a);
    l[c] = a;
  }
  float m = l[0];
  for (int c = 1; c < CC; c++) m = fmaxf(m, l[c]);
  float s = 0.0f;
  for (int c = 0; c < CC; c++) s += expf(l[c] - m);
  float lse = m + logf(s);
  for (int c = 0; c < CC; c++) out[bb * CC + c] = l[c] - lse;
}

// ---------------- launch ----------------
extern "C" void kernel_launch(void* const* d_in, const int* in_sizes, int n_in,
                              void* d_out, int out_size, void* d_ws, size_t ws_size,
                              hipStream_t stream) {
  (void)in_sizes; (void)n_in; (void)out_size; (void)ws_size;
  const float* x        = (const float*)d_in[0];
  const float* pos      = (const float*)d_in[1];
  const float* eattr    = (const float*)d_in[2];
  const float* w1a      = (const float*)d_in[3];
  const float* b1a      = (const float*)d_in[4];
  const float* w1b      = (const float*)d_in[5];
  const float* b1b      = (const float*)d_in[6];
  const float* root1    = (const float*)d_in[7];
  const float* bias1    = (const float*)d_in[8];
  const float* w2a      = (const float*)d_in[9];
  const float* b2a      = (const float*)d_in[10];
  const float* w2b      = (const float*)d_in[11];
  const float* b2b      = (const float*)d_in[12];
  const float* root2    = (const float*)d_in[13];
  const float* bias2    = (const float*)d_in[14];
  const float* fc1_w    = (const float*)d_in[15];
  const float* fc1_b    = (const float*)d_in[16];
  const float* fc2_w    = (const float*)d_in[17];
  const float* fc2_b    = (const float*)d_in[18];
  const int*   eidx     = (const int*)d_in[19];
  const int*   cluster1 = (const int*)d_in[20];
  const int*   cluster2 = (const int*)d_in[21];
  const int*   batch    = (const int*)d_in[22];
  float* out = (float*)d_out;

  // workspace carve-up (256B aligned)
  char* base = (char*)d_ws;
  size_t off = 0;
  auto carve = [&](size_t bytes) { char* p = base + off; off = (off + bytes + 255) & ~(size_t)255; return p; };
  float* s1      = (float*)carve((size_t)NN * HH1 * 4);
  float* cnt1    = (float*)carve((size_t)NN * 4);
  float* x1      = (float*)carve((size_t)NN * HH1 * 4);
  float* x_p     = (float*)carve((size_t)NN1 * HH1 * 4);
  float* pos_sum = (float*)carve((size_t)NN1 * 2 * 4);
  float* cntp    = (float*)carve((size_t)NN1 * 4);
  float* pos_p   = (float*)carve((size_t)NN1 * 2 * 4);
  float* maxabs  = (float*)carve(256);
  float* s2      = (float*)carve((size_t)NN1 * HH2 * 4);
  float* cnt2    = (float*)carve((size_t)NN1 * 4);
  float* x2      = (float*)carve((size_t)NN1 * HH2 * 4);
  float* x_p2    = (float*)carve((size_t)NN2 * HH2 * 4);
  int*   batch2  = (int*)carve((size_t)NN1 * 4);
  int*   batch3  = (int*)carve((size_t)NN2 * 4);
  float* g       = (float*)carve((size_t)BB * HH2 * 4);
  float* gc      = (float*)carve((size_t)BB * 4);
  float* hfc     = (float*)carve((size_t)BB * 128 * 4);
  _Float16* b1h  = (_Float16*)carve((size_t)K1PAD * HH1 * 2);
  _Float16* b2h  = (_Float16*)carve((size_t)K2PAD * HH2 * 2);
  _Float16* bf1h = (_Float16*)carve((size_t)64 * 128 * 2);

  auto blk = [](int n) { return (n + 255) / 256; };

  // init
  fill_f32<<<blk(NN * HH1), 256, 0, stream>>>(s1, 0.0f, NN * HH1);
  fill_f32<<<blk(NN), 256, 0, stream>>>(cnt1, 0.0f, NN);
  fill_f32<<<blk(NN1 * HH1), 256, 0, stream>>>(x_p, -1e30f, NN1 * HH1);
  fill_f32<<<blk(NN1 * 2), 256, 0, stream>>>(pos_sum, 0.0f, NN1 * 2);
  fill_f32<<<blk(NN1), 256, 0, stream>>>(cntp, 0.0f, NN1);
  fill_i32<<<1, 32, 0, stream>>>((int*)maxabs, 0, 1);
  fill_f32<<<blk(NN1 * HH2), 256, 0, stream>>>(s2, 0.0f, NN1 * HH2);
  fill_f32<<<blk(NN1), 256, 0, stream>>>(cnt2, 0.0f, NN1);
  fill_f32<<<blk(NN2 * HH2), 256, 0, stream>>>(x_p2, -1e30f, NN2 * HH2);
  fill_i32<<<blk(NN1), 256, 0, stream>>>(batch2, 0, NN1);
  fill_i32<<<blk(NN2), 256, 0, stream>>>(batch3, 0, NN2);
  fill_f32<<<blk(BB * HH2), 256, 0, stream>>>(g, 0.0f, BB * HH2);
  fill_f32<<<1, 64, 0, stream>>>(gc, 0.0f, BB);

  // B matrices -> f16, pre-swizzled into WMMA fragment order
  makeBfrag<FIN, HH1, K1PAD><<<blk(K1PAD * HH1), 256, 0, stream>>>(w1b, b1b, b1h);
  makeBfrag<HH1, HH2, K2PAD><<<blk(K2PAD * HH2), 256, 0, stream>>>(w2b, b2b, b2h);
  makeBfrag_fc1<<<blk(64 * 128), 256, 0, stream>>>(fc1_w, bf1h);

  // conv1 (WMMA) + node update + ELU
  nnconv_msg<FIN, HH1, K1PAD, false><<<EE / 16, 32, 0, stream>>>(
      x, eattr, w1a, b1a, eidx, cluster1, maxabs, b1h, s1, cnt1);
  node_update<FIN, HH1><<<blk(NN * HH1), 256, 0, stream>>>(s1, cnt1, x, root1, bias1, x1, NN);

  // graclus pool 1
  pool1<<<blk(NN * HH1), 256, 0, stream>>>(x1, pos, cluster1, batch, x_p, pos_sum, cntp, batch2);
  posdiv<<<blk(NN1), 256, 0, stream>>>(pos_sum, cntp, pos_p);
  cart_maxabs<<<blk(EE), 256, 0, stream>>>(pos_p, eidx, cluster1, (unsigned int*)maxabs);

  // conv2 (WMMA) + node update + ELU
  nnconv_msg<HH1, HH2, K2PAD, true><<<EE / 16, 32, 0, stream>>>(
      x_p, pos_p, w2a, b2a, eidx, cluster1, maxabs, b2h, s2, cnt2);
  node_update<HH1, HH2><<<blk(NN1 * HH2), 256, 0, stream>>>(s2, cnt2, x_p, root2, bias2, x2, NN1);

  // pool 2 + global mean pool
  pool2<<<blk(NN1 * HH2), 256, 0, stream>>>(x2, cluster2, batch2, x_p2, batch3);
  gpool<<<blk(NN2 * HH2), 256, 0, stream>>>(x_p2, batch3, g, gc);
  gdiv<<<blk(BB * HH2), 256, 0, stream>>>(g, gc);

  // head: fc1 via WMMA, fc2 + log_softmax scalar
  fc1_wmma<<<1, 128, 0, stream>>>(g, bf1h, fc1_b, hfc);
  fc2k<<<1, 64, 0, stream>>>(hfc, fc2_w, fc2_b, out);
}
